// SelfAttention_20882130993448
// MI455X (gfx1250) — matrix operand order repaired and checked by reading, not compiled, not run on hardware
//
#include <hip/hip_runtime.h>
#include <hip/hip_bf16.h>
#include <math.h>

typedef _Float16 v16h __attribute__((ext_vector_type(16)));
typedef _Float16 v8h  __attribute__((ext_vector_type(8)));
typedef float    v8f  __attribute__((ext_vector_type(8)));

#define B_  2
#define N_  4096
#define D_  512
#define H_  8
#define C_  64
#define TOK (B_*N_)   /* 8192 tokens */

// LDS tile strides (f16 elements), padded +8 to spread banks (keeps 16B align)
#define KSTR 72   /* 64 + 8 */
#define VSTR 40   /* 32 + 8 */
#define PSTR 40   /* 32 + 8 */

// ---------------------------------------------------------------------------
// WMMA helper: D = A(16x32 f16) * B(32x16 f16) + C(16x16 f32)
// ---------------------------------------------------------------------------
__device__ __forceinline__ v8f wmma32(v16h a, v16h b, v8f c) {
  return __builtin_amdgcn_wmma_f32_16x16x32_f16(
      /*neg_a=*/false, a, /*neg_b=*/false, b,
      /*c_mod=*/(short)0, c, /*reuse_a=*/false, /*reuse_b=*/false);
}

// A-fragment (16x32, MxK): lane = row; halfgroup lg picks k-phase.
// elements 0..15 <-> k = lg*16 + e
__device__ __forceinline__ v16h load_a_frag(const _Float16* rowp, int lg) {
  v8h c0 = *(const v8h*)(rowp + lg * 16);
  v8h c1 = *(const v8h*)(rowp + lg * 16 + 8);
  v16h a;
#pragma unroll
  for (int e = 0; e < 8; ++e) { a[e] = c0[e]; a[e + 8] = c1[e]; }
  return a;
}

// B-fragment (32x16, KxN): lane = column; 16 contiguous f16 per lane.
__device__ __forceinline__ v16h load_b_frag(const _Float16* p) {
  v8h c0 = *(const v8h*)(p);
  v8h c1 = *(const v8h*)(p + 8);
  v16h b;
#pragma unroll
  for (int e = 0; e < 8; ++e) { b[e] = c0[e]; b[e + 8] = c1[e]; }
  return b;
}

// ---------------------------------------------------------------------------
__global__ void cast_f16(const float* __restrict__ in, _Float16* __restrict__ out, int n) {
  int i = blockIdx.x * blockDim.x + threadIdx.x;
  if (i < n) out[i] = (_Float16)in[i];
}

__global__ void transpose_cast(const float* __restrict__ in, _Float16* __restrict__ out) {
  int n = blockIdx.x * 16 + threadIdx.x;
  int k = blockIdx.y * 16 + threadIdx.y;
  out[(size_t)n * D_ + k] = (_Float16)in[(size_t)k * D_ + n];
}

// ---------------------------------------------------------------------------
// Projection GEMM, 32 rows x 64 cols (one head) per wave.
// Wt stored transposed [outcol][k]. vtrans=1 stores V as [b][h][c][n].
// ---------------------------------------------------------------------------
__global__ void __launch_bounds__(32)
proj_gemm(const _Float16* __restrict__ xh, const _Float16* __restrict__ Wt,
          _Float16* __restrict__ outp, float scale, int vtrans) {
  int mt = blockIdx.x, h = blockIdx.y;
  int lane = threadIdx.x & 31, lg = lane >> 4, lm = lane & 15;
  const _Float16* arow0 = xh + (size_t)(mt * 32 + lm) * D_;
  const _Float16* arow1 = arow0 + (size_t)16 * D_;
  v8f acc[2][4] = {};
#pragma unroll 2
  for (int kk = 0; kk < D_; kk += 32) {
    v16h a0 = load_a_frag(arow0 + kk, lg);
    v16h a1 = load_a_frag(arow1 + kk, lg);
#pragma unroll
    for (int nt = 0; nt < 4; ++nt) {
      int ncol = h * C_ + nt * 16 + lm;
      v16h b = load_b_frag(Wt + (size_t)ncol * D_ + kk + lg * 16);
      acc[0][nt] = wmma32(a0, b, acc[0][nt]);
      acc[1][nt] = wmma32(a1, b, acc[1][nt]);
    }
  }
#pragma unroll
  for (int rh = 0; rh < 2; ++rh)
#pragma unroll
    for (int nt = 0; nt < 4; ++nt)
#pragma unroll
      for (int r = 0; r < 8; ++r) {
        int m = rh * 16 + lg * 8 + r;
        int t = mt * 32 + m;
        int b = t >> 12, n = t & (N_ - 1);
        int col = nt * 16 + lm;
        _Float16 v = (_Float16)(acc[rh][nt][r] * scale);
        if (!vtrans)
          outp[((size_t)(b * H_ + h) * N_ + n) * C_ + col] = v;   // [b][h][n][c]
        else
          outp[((size_t)(b * H_ + h) * C_ + col) * N_ + n] = v;   // [b][h][c][n]
      }
}

// ---------------------------------------------------------------------------
// Flash attention: 4 waves/WG, 32 query rows/wave (128 rows per WG).
// Each 32-key block of K (32x64) and V^T (64x32) is staged once into LDS
// by the whole WG; every wave then does 8 QK^T WMMAs + 8 PV WMMAs from LDS.
// ---------------------------------------------------------------------------
__global__ void __launch_bounds__(128)
attn_fwd(const _Float16* __restrict__ Qs, const _Float16* __restrict__ Kf,
         const _Float16* __restrict__ Vt, _Float16* __restrict__ Oh) {
  int qt = blockIdx.x, h = blockIdx.y, b = blockIdx.z;
  int tid = threadIdx.x;
  int w = tid >> 5;
  int lane = tid & 31, lg = lane >> 4, lm = lane & 15;
  size_t bh = (size_t)(b * H_ + h);
  int rowbase = qt * 128 + w * 32;
  const _Float16* Qb = Qs + (bh * N_ + rowbase) * C_;
  const _Float16* Kb = Kf + bh * N_ * C_;
  const _Float16* Vb = Vt + bh * C_ * N_;

  __shared__ __align__(16) _Float16 Ktile[32 * KSTR];
  __shared__ __align__(16) _Float16 Vtile[64 * VSTR];
  __shared__ __align__(16) _Float16 Plds[4][32 * PSTR];

  // Q fragments: 2 row-halves x 2 k-phases (channels 0..31, 32..63)
  v16h aq[2][2];
#pragma unroll
  for (int rh = 0; rh < 2; ++rh)
#pragma unroll
    for (int f = 0; f < 2; ++f)
      aq[rh][f] = load_a_frag(Qb + (size_t)(rh * 16 + lm) * C_ + f * 32, lg);

  float mrow[2][8], lrow[2][8];
#pragma unroll
  for (int rh = 0; rh < 2; ++rh)
#pragma unroll
    for (int r = 0; r < 8; ++r) { mrow[rh][r] = -3.0e38f; lrow[rh][r] = 0.f; }
  v8f acc[2][4] = {};

  for (int j0 = 0; j0 < N_; j0 += 32) {
    __syncthreads();  // previous block's LDS reads done before overwrite
    {
      // Stage K block: 32 keys x 64 ch; thread -> (key row, 16-ch group)
      int kr = tid >> 2, cg = tid & 3;
      const v8h* ks = (const v8h*)(Kb + (size_t)(j0 + kr) * C_ + cg * 16);
      v8h k0 = ks[0], k1 = ks[1];
      *(v8h*)(&Ktile[kr * KSTR + cg * 16]) = k0;
      *(v8h*)(&Ktile[kr * KSTR + cg * 16 + 8]) = k1;
      // Stage V^T block: 64 ch x 32 keys; thread -> (channel, 16-key half)
      int c = tid >> 1, hf = tid & 1;
      const v8h* vs = (const v8h*)(Vb + (size_t)c * N_ + j0 + hf * 16);
      v8h v0 = vs[0], v1 = vs[1];
      *(v8h*)(&Vtile[c * VSTR + hf * 16]) = v0;
      *(v8h*)(&Vtile[c * VSTR + hf * 16 + 8]) = v1;
    }
    __syncthreads();  // tiles visible to all waves

#pragma unroll
    for (int rh = 0; rh < 2; ++rh) {
      // S = Q * K^T for this row-half: 2 key sub-tiles x 2 k-phases
      v8f S[2] = {{0}, {0}};
#pragma unroll
      for (int s = 0; s < 2; ++s) {
        v16h bk0 = load_b_frag(&Ktile[(s * 16 + lm) * KSTR + lg * 16]);
        v16h bk1 = load_b_frag(&Ktile[(s * 16 + lm) * KSTR + 32 + lg * 16]);
        S[s] = wmma32(aq[rh][0], bk0, S[s]);
        S[s] = wmma32(aq[rh][1], bk1, S[s]);
      }
      // Online softmax: row M = lg*8+r lives in VGPR r across one 16-lane half
      float corr[8];
#pragma unroll
      for (int r = 0; r < 8; ++r) {
        float mx = fmaxf(S[0][r], S[1][r]);
#pragma unroll
        for (int msk = 8; msk >= 1; msk >>= 1) mx = fmaxf(mx, __shfl_xor(mx, msk, 32));
        float mnew = fmaxf(mrow[rh][r], mx);
        float c  = __expf(mrow[rh][r] - mnew);
        float p0 = __expf(S[0][r] - mnew);
        float p1 = __expf(S[1][r] - mnew);
        S[0][r] = p0; S[1][r] = p1;
        float rs = p0 + p1;
#pragma unroll
        for (int msk = 8; msk >= 1; msk >>= 1) rs += __shfl_xor(rs, msk, 32);
        lrow[rh][r] = lrow[rh][r] * c + rs;
        mrow[rh][r] = mnew;
        corr[r] = c;
      }
#pragma unroll
      for (int nt = 0; nt < 4; ++nt)
#pragma unroll
        for (int r = 0; r < 8; ++r) acc[rh][nt][r] *= corr[r];

      // P: C-layout -> per-wave LDS region -> A-layout (same-wave DS is in-order)
#pragma unroll
      for (int r = 0; r < 8; ++r) {
        int m = rh * 16 + lg * 8 + r;
        Plds[w][m * PSTR + lm]      = (_Float16)S[0][r];
        Plds[w][m * PSTR + 16 + lm] = (_Float16)S[1][r];
      }
    }
    // PV: acc += P * V for both row-halves
#pragma unroll
    for (int rh = 0; rh < 2; ++rh) {
      int m = rh * 16 + lm;
      v16h ap;
      {
        v8h c0 = *(const v8h*)(&Plds[w][m * PSTR + lg * 16]);
        v8h c1 = *(const v8h*)(&Plds[w][m * PSTR + lg * 16 + 8]);
#pragma unroll
        for (int e = 0; e < 8; ++e) { ap[e] = c0[e]; ap[e + 8] = c1[e]; }
      }
#pragma unroll
      for (int nt = 0; nt < 4; ++nt) {
        v16h bv = load_b_frag(&Vtile[(nt * 16 + lm) * VSTR + lg * 16]);
        acc[rh][nt] = wmma32(ap, bv, acc[rh][nt]);
      }
    }
  }

  // Epilogue: divide by row sum, store merged-heads f16
#pragma unroll
  for (int rh = 0; rh < 2; ++rh)
#pragma unroll
    for (int nt = 0; nt < 4; ++nt)
#pragma unroll
      for (int r = 0; r < 8; ++r) {
        int m = rh * 16 + lg * 8 + r;
        float val = acc[rh][nt][r] / lrow[rh][r];
        Oh[((size_t)(b * N_) + rowbase + m) * (H_ * C_) + h * C_ + nt * 16 + lm] =
            (_Float16)val;
      }
}

// ---------------------------------------------------------------------------
// Output projection: out[8192x512] f32 = Ah * Wo + bo ; 32 rows/wave.
// ---------------------------------------------------------------------------
__global__ void __launch_bounds__(32)
out_gemm(const _Float16* __restrict__ Ah, const _Float16* __restrict__ Wot,
         const float* __restrict__ bo, float* __restrict__ outp) {
  int mt = blockIdx.x, g = blockIdx.y;  // g = column group of 64
  int lane = threadIdx.x & 31, lg = lane >> 4, lm = lane & 15;
  const _Float16* arow0 = Ah + (size_t)(mt * 32 + lm) * D_;
  const _Float16* arow1 = arow0 + (size_t)16 * D_;
  v8f acc[2][4] = {};
#pragma unroll 2
  for (int kk = 0; kk < D_; kk += 32) {
    v16h a0 = load_a_frag(arow0 + kk, lg);
    v16h a1 = load_a_frag(arow1 + kk, lg);
#pragma unroll
    for (int nt = 0; nt < 4; ++nt) {
      int ncol = g * 64 + nt * 16 + lm;
      v16h bfrag = load_b_frag(Wot + (size_t)ncol * D_ + kk + lg * 16);
      acc[0][nt] = wmma32(a0, bfrag, acc[0][nt]);
      acc[1][nt] = wmma32(a1, bfrag, acc[1][nt]);
    }
  }
#pragma unroll
  for (int rh = 0; rh < 2; ++rh)
#pragma unroll
    for (int nt = 0; nt < 4; ++nt)
#pragma unroll
      for (int r = 0; r < 8; ++r) {
        int m = rh * 16 + lg * 8 + r;
        int t = mt * 32 + m;
        int col = g * 64 + nt * 16 + lm;
        outp[(size_t)t * D_ + col] = acc[rh][nt][r] + bo[col];
      }
}

// ---------------------------------------------------------------------------
extern "C" void kernel_launch(void* const* d_in, const int* in_sizes, int n_in,
                              void* d_out, int out_size, void* d_ws, size_t ws_size,
                              hipStream_t stream) {
  (void)in_sizes; (void)n_in; (void)out_size; (void)ws_size;
  const float* x  = (const float*)d_in[0];
  const float* Wq = (const float*)d_in[1];
  const float* Wk = (const float*)d_in[2];
  const float* Wv = (const float*)d_in[3];
  const float* Wo = (const float*)d_in[4];
  const float* bo = (const float*)d_in[5];
  float* outp = (float*)d_out;

  // Workspace carve-up (f16 elements): ~42 MB total
  _Float16* ws = (_Float16*)d_ws;
  size_t off = 0;
  _Float16* xh  = ws + off; off += (size_t)TOK * D_;   // x as f16
  _Float16* Wqt = ws + off; off += (size_t)D_ * D_;    // W^T f16
  _Float16* Wkt = ws + off; off += (size_t)D_ * D_;
  _Float16* Wvt = ws + off; off += (size_t)D_ * D_;
  _Float16* Wot = ws + off; off += (size_t)D_ * D_;
  _Float16* Qs  = ws + off; off += (size_t)TOK * D_;   // [b][h][n][c], pre-scaled
  _Float16* Kf  = ws + off; off += (size_t)TOK * D_;   // [b][h][n][c]
  _Float16* Vt  = ws + off; off += (size_t)TOK * D_;   // [b][h][c][n]
  _Float16* Ah  = ws + off; off += (size_t)TOK * D_;   // attention out, merged heads

  cast_f16<<<(TOK * D_ + 255) / 256, 256, 0, stream>>>(x, xh, TOK * D_);
  dim3 tg(32, 32), tb(16, 16);
  transpose_cast<<<tg, tb, 0, stream>>>(Wq, Wqt);
  transpose_cast<<<tg, tb, 0, stream>>>(Wk, Wkt);
  transpose_cast<<<tg, tb, 0, stream>>>(Wv, Wvt);
  transpose_cast<<<tg, tb, 0, stream>>>(Wo, Wot);

  dim3 pg(TOK / 32, H_);
  proj_gemm<<<pg, 32, 0, stream>>>(xh, Wqt, Qs, 0.125f, 0);  // scale = C^-0.5
  proj_gemm<<<pg, 32, 0, stream>>>(xh, Wkt, Kf, 1.0f, 0);
  proj_gemm<<<pg, 32, 0, stream>>>(xh, Wvt, Vt, 1.0f, 1);

  dim3 ag(N_ / 128, H_, B_);
  attn_fwd<<<ag, 128, 0, stream>>>(Qs, Kf, Vt, Ah);

  dim3 og(TOK / 32, D_ / 64);
  out_gemm<<<og, 32, 0, stream>>>(Ah, Wot, bo, outp);
}
